// HashingMemory_8529805050327
// MI455X (gfx1250) — matrix operand order repaired and checked by reading, not compiled, not run on hardware
//
#include <hip/hip_runtime.h>
#include <math.h>

// ---------------------------------------------------------------- types ----
typedef __attribute__((ext_vector_type(16))) __bf16 v16bf;
typedef __attribute__((ext_vector_type(8)))  float  v8f;

union AFrag { unsigned int u[8]; v16bf v; };
union BFrag { uint4 q[2];        v16bf v; };

__device__ __forceinline__ unsigned short f32_to_bf16(float f) {
  unsigned int u = __float_as_uint(f);
  u += 0x7fffu + ((u >> 16) & 1u);   // round to nearest even
  return (unsigned short)(u >> 16);
}

// async global->LDS 16B copy (CDNA5, tracked by ASYNCcnt)
__device__ __forceinline__ void async_b128(unsigned lds_off,
                                           const unsigned short* g) {
  unsigned long long ga = (unsigned long long)(uintptr_t)g;
  asm volatile("global_load_async_to_lds_b128 %0, %1, off"
               :: "v"(lds_off), "v"(ga) : "memory");
}
__device__ __forceinline__ void wait_async0() {
  asm volatile("s_wait_asynccnt 0x0" ::: "memory");
}

// ------------------------------------------------------------- constants ---
#define HEADS 4
#define K_DIM 512
#define HALF  256
#define N_KEYS 512
#define KNN   32
#define D_IN  512
#define D_OUT 512
#define NTOK  2048          // 2*1024
#define NSCORE 4096         // HEADS*2*N_KEYS

// ------------------------------------------------------ f32 -> bf16 cast ---
__global__ void pkm_to_bf16(unsigned short* __restrict__ dst,
                            const float* __restrict__ src, int n) {
  int i = blockIdx.x * 256 + threadIdx.x;
  if (i < n) dst[i] = f32_to_bf16(src[i]);
}

// ---------------------------------------------- fold keys into Wq (bf16) ---
// Ws[r, d] = sum_j keys[r, j] * Wq[wqbase(r) + j, d],  r = (h, half, nk)
__global__ __launch_bounds__(256) void pkm_build_ws(
    unsigned short* __restrict__ Wsb,
    const float* __restrict__ keys, const float* __restrict__ Wq) {
  __shared__ float sk[16][17];
  __shared__ float sw[16][17];
  int r0 = blockIdx.y * 16, d0 = blockIdx.x * 16;
  int ty = threadIdx.y, tx = threadIdx.x;
  int h    = r0 >> 10;          // r / 1024
  int half = (r0 >> 9) & 1;     // (r/512) % 2
  int wqbase = h * K_DIM + half * HALF;
  int r = r0 + ty, d = d0 + tx;
  float acc = 0.f;
  for (int j0 = 0; j0 < HALF; j0 += 16) {
    sk[ty][tx] = keys[(size_t)r * HALF + j0 + tx];
    sw[ty][tx] = Wq[(size_t)(wqbase + j0 + ty) * D_IN + d];
    __syncthreads();
#pragma unroll
    for (int j = 0; j < 16; j++) acc += sk[ty][j] * sw[j][tx];
    __syncthreads();
  }
  Wsb[(size_t)r * D_IN + d] = f32_to_bf16(acc);
}

// score bias: bs[r] = keys[r,:] . bq[slice]
__global__ void pkm_build_bs(float* __restrict__ bs,
                             const float* __restrict__ keys,
                             const float* __restrict__ bq) {
  int r = blockIdx.x * 256 + threadIdx.x;
  if (r >= NSCORE) return;
  int h = r >> 10, half = (r >> 9) & 1;
  const float* kk = keys + (size_t)r * HALF;
  const float* bb = bq + h * K_DIM + half * HALF;
  float acc = 0.f;
  for (int j = 0; j < HALF; j++) acc += kk[j] * bb[j];
  bs[r] = acc;
}

// ----------------------------------------------- bf16 WMMA GEMM, C=A*B^T ---
// C (MxN, f32) = A (MxK bf16 row-major) * B (NxK bf16 row-major)^T + bias[n]
// Double-buffered LDS tiles filled with global_load_async_to_lds_b128.
#define BM 128
#define BN 128
#define BK 32
#define LDT 40     // padded LDS stride in halfs (80B, 16B aligned)

__global__ __launch_bounds__(256) void pkm_gemm_bt(
    float* __restrict__ C, const unsigned short* __restrict__ A,
    const unsigned short* __restrict__ B, const float* __restrict__ bias,
    int M, int N, int K) {
  __shared__ unsigned short sA[2][BM * LDT];
  __shared__ unsigned short sB[2][BN * LDT];
  int tid  = threadIdx.x;
  int wave = tid >> 5, lane = tid & 31;
  int wm = wave & 3, wn = wave >> 2;       // 4x2 wave grid, 32x64 per wave
  int g  = lane >> 4, lr = lane & 15;
  int block_m = blockIdx.y * BM, block_n = blockIdx.x * BN;

  v8f acc[2][4];
#pragma unroll
  for (int i = 0; i < 2; i++)
#pragma unroll
    for (int j = 0; j < 4; j++)
#pragma unroll
      for (int r = 0; r < 8; r++) acc[i][j][r] = 0.f;

  // staging assignment: each thread owns 16 halfs (two b128) per matrix
  int ldrow = tid >> 1;
  int ldcol = (tid & 1) * 16;
  const unsigned short* gA = A + (size_t)(block_m + ldrow) * K + ldcol;
  const unsigned short* gB = B + (size_t)(block_n + ldrow) * K + ldcol;
  unsigned la[2], lb[2];
  la[0] = (unsigned)(uintptr_t)&sA[0][ldrow * LDT + ldcol];
  la[1] = (unsigned)(uintptr_t)&sA[1][ldrow * LDT + ldcol];
  lb[0] = (unsigned)(uintptr_t)&sB[0][ldrow * LDT + ldcol];
  lb[1] = (unsigned)(uintptr_t)&sB[1][ldrow * LDT + ldcol];

  const int nIter = K / BK;

  // prologue: stage tile 0 into buffer 0
  async_b128(la[0],      gA);
  async_b128(la[0] + 16, gA + 8);
  async_b128(lb[0],      gB);
  async_b128(lb[0] + 16, gB + 8);
  wait_async0();
  __syncthreads();

  for (int it = 0; it < nIter; ++it) {
    int cur = it & 1;
    if (it + 1 < nIter) {      // stage next tile into other buffer (async)
      int nxt = cur ^ 1;
      const unsigned short* pa = gA + (it + 1) * BK;
      const unsigned short* pb = gB + (it + 1) * BK;
      async_b128(la[nxt],      pa);
      async_b128(la[nxt] + 16, pa + 8);
      async_b128(lb[nxt],      pb);
      async_b128(lb[nxt] + 16, pb + 8);
    }

    AFrag af[2];
#pragma unroll
    for (int i = 0; i < 2; i++) {
      const unsigned short* base = &sA[cur][(wm * 32 + i * 16 + lr) * LDT];
      // 16-bit A layout: VGPR v: k=16g+2v+p (contiguous, matches B)
#pragma unroll
      for (int v = 0; v < 4; v++) {
        af[i].u[v]     = *(const unsigned int*)(base + 16 * g + 2 * v);
        af[i].u[v + 4] = *(const unsigned int*)(base + 16 * g + 8 + 2 * v);
      }
    }
    BFrag bf[4];
#pragma unroll
    for (int j = 0; j < 4; j++) {
      // 16-bit B layout: lanes 0-15 hold k=0..15, lanes 16-31 hold k=16..31
      const uint4* p = (const uint4*)&sB[cur][(wn * 64 + j * 16 + lr) * LDT + 16 * g];
      bf[j].q[0] = p[0]; bf[j].q[1] = p[1];
    }
#pragma unroll
    for (int i = 0; i < 2; i++)
#pragma unroll
      for (int j = 0; j < 4; j++)
        acc[i][j] = __builtin_amdgcn_wmma_f32_16x16x32_bf16(
            false, af[i].v, false, bf[j].v, (short)0, acc[i][j], false, false);

    wait_async0();     // next tile landed (this wave's share)
    __syncthreads();   // all waves' shares landed; cur buffer free for reuse
  }

#pragma unroll
  for (int i = 0; i < 2; i++) {
#pragma unroll
    for (int j = 0; j < 4; j++) {
      int n = block_n + wn * 64 + j * 16 + lr;
      float bb = bias ? bias[n] : 0.f;
#pragma unroll
      for (int r = 0; r < 8; r++) {
        int m = block_m + wm * 32 + i * 16 + r + 8 * g;   // C/D layout
        C[(size_t)m * N + n] = acc[i][j][r] + bb;
      }
    }
  }
}

// ---------------------------------------------------- product-key top-k ----
// one wave32 per (token, head)
__global__ __launch_bounds__(32) void pkm_topk(
    int* __restrict__ idx_out, float* __restrict__ p_out,
    const float* __restrict__ S) {
  int n = blockIdx.x, h = blockIdx.y, lane = threadIdx.x;
  const float* s1 = S + (size_t)n * NSCORE + h * (2 * N_KEYS);
  const float* s2 = s1 + N_KEYS;

  float v1[16], v2[16];
#pragma unroll
  for (int j = 0; j < 16; j++) {
    v1[j] = s1[lane * 16 + j];
    v2[j] = s2[lane * 16 + j];
  }
  const float NEG = -__builtin_inff();
  float ts1 = NEG, ts2 = NEG;           // lane r keeps rank-r result
  int   ti1 = 0,   ti2 = 0;

  for (int r = 0; r < KNN; r++) {       // top-32 of s1
    float bv = v1[0]; int bj = 0;
#pragma unroll
    for (int j = 1; j < 16; j++) if (v1[j] > bv) { bv = v1[j]; bj = j; }
    int bidx = lane * 16 + bj;
    for (int off = 16; off > 0; off >>= 1) {
      float ov = __shfl_xor(bv, off); int oi = __shfl_xor(bidx, off);
      if (ov > bv || (ov == bv && oi < bidx)) { bv = ov; bidx = oi; }
    }
    if (lane == r) { ts1 = bv; ti1 = bidx; }
    int loc = bidx - lane * 16;
#pragma unroll
    for (int j = 0; j < 16; j++) if (loc == j) v1[j] = NEG;
  }
  for (int r = 0; r < KNN; r++) {       // top-32 of s2
    float bv = v2[0]; int bj = 0;
#pragma unroll
    for (int j = 1; j < 16; j++) if (v2[j] > bv) { bv = v2[j]; bj = j; }
    int bidx = lane * 16 + bj;
    for (int off = 16; off > 0; off >>= 1) {
      float ov = __shfl_xor(bv, off); int oi = __shfl_xor(bidx, off);
      if (ov > bv || (ov == bv && oi < bidx)) { bv = ov; bidx = oi; }
    }
    if (lane == r) { ts2 = bv; ti2 = bidx; }
    int loc = bidx - lane * 16;
#pragma unroll
    for (int j = 0; j < 16; j++) if (loc == j) v2[j] = NEG;
  }

  // cartesian 32x32: lane i owns row i -> cand[j] = ts1[i] + ts2[j]
  float cand[KNN];
#pragma unroll
  for (int j = 0; j < KNN; j++) cand[j] = ts1 + __shfl(ts2, j);

  float sc = NEG; int ci = 0;
  for (int r = 0; r < KNN; r++) {       // top-32 of 1024
    float bv = cand[0]; int bj = 0;
#pragma unroll
    for (int j = 1; j < KNN; j++) if (cand[j] > bv) { bv = cand[j]; bj = j; }
    int bidx = lane * KNN + bj;
    for (int off = 16; off > 0; off >>= 1) {
      float ov = __shfl_xor(bv, off); int oi = __shfl_xor(bidx, off);
      if (ov > bv || (ov == bv && oi < bidx)) { bv = ov; bidx = oi; }
    }
    if (lane == r) { sc = bv; ci = bidx; }
    int loc = bidx - lane * KNN;
#pragma unroll
    for (int j = 0; j < KNN; j++) if (loc == j) cand[j] = NEG;
  }

  int i1 = __shfl(ti1, ci >> 5);
  int i2 = __shfl(ti2, ci & 31);
  int index = i1 * N_KEYS + i2;

  // softmax over the 32 selected scores (one per lane)
  float m = sc;
  for (int off = 16; off > 0; off >>= 1) m = fmaxf(m, __shfl_xor(m, off));
  float e = __expf(sc - m);
  float ssum = e;
  for (int off = 16; off > 0; off >>= 1) ssum += __shfl_xor(ssum, off);
  float p = e / ssum;

  size_t o = ((size_t)n * HEADS + h) * KNN + lane;
  idx_out[o] = index;
  p_out[o]   = p;
}

// ------------------------------------------------ weighted embedding bag ---
__global__ __launch_bounds__(128) void pkm_bag(
    float* __restrict__ bag, const float* __restrict__ values,
    const int* __restrict__ idx, const float* __restrict__ p) {
  __shared__ int   sidx[HEADS * KNN];
  __shared__ float sp[HEADS * KNN];
  int n = blockIdx.x, t = threadIdx.x;
  sidx[t] = idx[(size_t)n * (HEADS * KNN) + t];
  sp[t]   = p[(size_t)n * (HEADS * KNN) + t];
  __syncthreads();
  int d = t * 4;
  float4 acc = {0.f, 0.f, 0.f, 0.f};
  for (int k = 0; k < HEADS * KNN; k++) {
    if (k + 1 < HEADS * KNN)
      __builtin_prefetch(values + (size_t)sidx[k + 1] * D_OUT + d, 0, 0);
    const float4 v = *(const float4*)(values + (size_t)sidx[k] * D_OUT + d);
    float w = sp[k];
    acc.x += w * v.x; acc.y += w * v.y; acc.z += w * v.z; acc.w += w * v.w;
  }
  *(float4*)(bag + (size_t)n * D_OUT + d) = acc;
}

// ----------------------------------- silu gate * bag -> bf16 for final GEMM
__global__ void pkm_silu_gate(unsigned short* __restrict__ gated,
                              const float* __restrict__ gatelin,
                              const float* __restrict__ bsw,
                              const float* __restrict__ bag, int total) {
  int i = blockIdx.x * 256 + threadIdx.x;
  if (i >= total) return;
  float g = gatelin[i] + bsw[i & (D_OUT - 1)];
  float s = g / (1.0f + __expf(-g));
  gated[i] = f32_to_bf16(bag[i] * s);
}

// ---------------------------------------------------------------- launch ---
extern "C" void kernel_launch(void* const* d_in, const int* in_sizes, int n_in,
                              void* d_out, int out_size, void* d_ws, size_t ws_size,
                              hipStream_t stream) {
  const float* x      = (const float*)d_in[0];
  const float* Wq     = (const float*)d_in[1];
  const float* bq     = (const float*)d_in[2];
  const float* keys   = (const float*)d_in[3];
  const float* values = (const float*)d_in[4];
  const float* Wsw    = (const float*)d_in[5];
  const float* bsw    = (const float*)d_in[6];
  const float* Wv     = (const float*)d_in[7];
  const float* bv     = (const float*)d_in[8];
  float* y = (float*)d_out;

  // workspace layout (256B aligned slabs)
  char* w = (char*)d_ws;
  size_t off = 0;
  auto alloc = [&](size_t bytes) { void* p = w + off; off = (off + bytes + 255) & ~(size_t)255; return p; };
  unsigned short* Xb      = (unsigned short*)alloc((size_t)NTOK * D_IN * 2);
  unsigned short* Wsb     = (unsigned short*)alloc((size_t)NSCORE * D_IN * 2);
  unsigned short* Wswb    = (unsigned short*)alloc((size_t)D_OUT * D_IN * 2);
  unsigned short* Wvb     = (unsigned short*)alloc((size_t)D_OUT * D_OUT * 2);
  float*          bs      = (float*)alloc((size_t)NSCORE * 4);
  int*            idx     = (int*)alloc((size_t)NTOK * HEADS * KNN * 4);
  float*          prob    = (float*)alloc((size_t)NTOK * HEADS * KNN * 4);
  float*          bag     = (float*)alloc((size_t)NTOK * D_OUT * 4);
  unsigned short* gated   = (unsigned short*)alloc((size_t)NTOK * D_OUT * 2);
  float*          gatelin = (float*)alloc((size_t)NTOK * D_OUT * 4);
  float*          S       = (float*)alloc((size_t)NTOK * NSCORE * 4);
  (void)ws_size; (void)n_in; (void)in_sizes; (void)out_size;

  // 1. bf16 conversions
  pkm_to_bf16<<<(NTOK * D_IN + 255) / 256, 256, 0, stream>>>(Xb, x, NTOK * D_IN);
  pkm_to_bf16<<<(D_OUT * D_IN + 255) / 256, 256, 0, stream>>>(Wswb, Wsw, D_OUT * D_IN);
  pkm_to_bf16<<<(D_OUT * D_OUT + 255) / 256, 256, 0, stream>>>(Wvb, Wv, D_OUT * D_OUT);

  // 2. fold keys into query weights: Ws (4096x512 bf16), bs (4096)
  pkm_build_ws<<<dim3(D_IN / 16, NSCORE / 16), dim3(16, 16), 0, stream>>>(Wsb, keys, Wq);
  pkm_build_bs<<<(NSCORE + 255) / 256, 256, 0, stream>>>(bs, keys, bq);

  // 3. all sub-key scores in one WMMA GEMM: S = x @ Ws^T + bs  (2048x4096)
  pkm_gemm_bt<<<dim3(NSCORE / BN, NTOK / BM), 256, 0, stream>>>(
      S, Xb, Wsb, bs, NTOK, NSCORE, D_IN);

  // 4. per-(token,head) product-key top-32 + softmax
  pkm_topk<<<dim3(NTOK, HEADS), 32, 0, stream>>>(idx, prob, S);

  // 5. weighted embedding bag (bandwidth-bound: ~512MB of gathers)
  pkm_bag<<<NTOK, 128, 0, stream>>>(bag, values, idx, prob);

  // 6. gate GEMM (WMMA) + silu * bag -> bf16
  pkm_gemm_bt<<<dim3(D_OUT / BN, NTOK / BM), 256, 0, stream>>>(
      gatelin, Xb, Wswb, nullptr, NTOK, D_OUT, D_IN);
  pkm_silu_gate<<<(NTOK * D_OUT + 255) / 256, 256, 0, stream>>>(
      gated, gatelin, bsw, bag, NTOK * D_OUT);

  // 7. output projection (WMMA): y = gated @ Wv^T + bv
  pkm_gemm_bt<<<dim3(D_OUT / BN, NTOK / BM), 256, 0, stream>>>(
      y, gated, Wvb, bv, NTOK, D_OUT, D_OUT);
}
